// cluster_tree2D_32117765439934
// MI455X (gfx1250) — matrix operand order repaired and checked by reading, not compiled, not run on hardware
//
#include <hip/hip_runtime.h>
#include <hip/hip_bf16.h>

// ---------------------------------------------------------------------------
// Types for CDNA5 WMMA (wave32, 16x16x32 bf16 -> f32)
// ---------------------------------------------------------------------------
typedef __attribute__((ext_vector_type(16))) __bf16 v16bf;
typedef __attribute__((ext_vector_type(8)))  float  v8f;

#define H_DIM 512
#define PF 516   // f32 LDS row pitch (elements)
#define PB 520   // bf16 LDS row pitch (elements)

__device__ __forceinline__ unsigned short f2bf(float f) {
    unsigned u = __float_as_uint(f);
    u += 0x7fffu + ((u >> 16) & 1u);   // round-to-nearest-even
    return (unsigned short)(u >> 16);
}

__device__ __forceinline__ float sigmoidf(float x) {
    return 1.0f / (1.0f + __expf(-x));
}

__device__ __forceinline__ v8f wmma_bf16(v16bf a, v16bf b, v8f c) {
    return __builtin_amdgcn_wmma_f32_16x16x32_bf16(
        /*neg_a=*/false, a, /*neg_b=*/false, b,
        /*c_mod=*/(short)0, c, /*reuse_a=*/false, /*reuse_b=*/false);
}

union TileU { uint4 q[2]; v16bf v; };

// A-matrix tile (16x32 bf16), ISA braided layout:
// lane half 0: V0-3 <- K 0..7,  V4-7 <- K 16..23
// lane half 1: V0-3 <- K 8..15, V4-7 <- K 24..31
// `row` = base pointer of this lane's row (M = lane&15).
__device__ __forceinline__ v16bf load_a(const unsigned short* row, int k0, int lh) {
    const unsigned short* p = row + k0 + lh * 8;
    TileU u;
    u.q[0] = *(const uint4*)(p);
    u.q[1] = *(const uint4*)(p + 16);
    return u.v;
}

// B-matrix tile (32x16 bf16), same braided layout as A:
// lane half 0: K 0..7, K 16..23; lane half 1: K 8..15, K 24..31.
// `row` = pointer to weight row (N = lane&15).
__device__ __forceinline__ v16bf load_b(const unsigned short* row, int k0, int lh) {
    const unsigned short* p = row + k0 + lh * 8;
    TileU u;
    u.q[0] = *(const uint4*)(p);
    u.q[1] = *(const uint4*)(p + 16);
    return u.v;
}

// ---------------------------------------------------------------------------
// f32 -> bf16 bulk convert
// ---------------------------------------------------------------------------
__global__ void cvt_f32_bf16(const float* __restrict__ src,
                             unsigned short* __restrict__ dst, int n) {
    for (int i = blockIdx.x * blockDim.x + threadIdx.x; i < n;
         i += gridDim.x * blockDim.x)
        dst[i] = f2bf(src[i]);
}

// ---------------------------------------------------------------------------
// Leaf layer: z = sigmoid(x @ Wz^T); h = (1-z) * tanh(x @ Wh^T)
// One block = 16 leaves, 8 waves, each wave does 4 N-tiles of 16.
// ---------------------------------------------------------------------------
__global__ __launch_bounds__(256)
void tree_gru_leaf(const unsigned short* __restrict__ xbf,
                   float* __restrict__ hstate,
                   const unsigned short* __restrict__ Wz_bf,
                   const unsigned short* __restrict__ Wh_bf,
                   int npt) {
    const int tid  = threadIdx.x;
    const int lane = tid & 31, wave = tid >> 5;
    const int lh = lane >> 4, ln = lane & 15;

    const int flatA = blockIdx.x * 16 + ln;            // leaf flat index (row M=ln)
    const int nodeA = (flatA >> 11) * npt + (flatA & 2047);
    const unsigned short* xrow = xbf + (long)nodeA * H_DIM;

#pragma unroll
    for (int t = 0; t < 4; ++t) {
        const int n0 = wave * 64 + t * 16;
        const unsigned short* wz = Wz_bf + (long)(n0 + ln) * H_DIM;
        const unsigned short* wh = Wh_bf + (long)(n0 + ln) * H_DIM;
        v8f za = {}, ha = {};
#pragma unroll
        for (int k0 = 0; k0 < H_DIM; k0 += 32) {
            v16bf ax = load_a(xrow, k0, lh);
            za = wmma_bf16(ax, load_b(wz, k0, lh), za);
            ha = wmma_bf16(ax, load_b(wh, k0, lh), ha);
        }
#pragma unroll
        for (int d = 0; d < 8; ++d) {
            const int m    = d + 8 * lh;
            const int flat = blockIdx.x * 16 + m;
            const int node = (flat >> 11) * npt + (flat & 2047);
            const float z  = sigmoidf(za[d]);
            hstate[(long)node * H_DIM + n0 + ln] = (1.0f - z) * tanhf(ha[d]);
        }
    }
}

// ---------------------------------------------------------------------------
// Fused GRU layer (bottom-up or top-down).
//   hh = sum of two children h (up) or parent hd (down)
//   z  = sig(x@Wz^T + hh@Uz^T + bz);  r = sig(x@Wr^T + hh@Ur^T + br)
//   cand = tanh(x@Wh^T + (r*hh)@Uh^T)
//   out  = z*hh + (1-z)*cand
// One block = 16 output nodes.
// ---------------------------------------------------------------------------
template <bool IS_DOWN>
__global__ __launch_bounds__(256)
void tree_gru_layer(const unsigned short* __restrict__ xbf,
                    float* __restrict__ hstate,
                    const unsigned short* __restrict__ Wz_bf,
                    const unsigned short* __restrict__ Uz_bf,
                    const float* __restrict__ bz,
                    const unsigned short* __restrict__ Wr_bf,
                    const unsigned short* __restrict__ Ur_bf,
                    const float* __restrict__ br,
                    const unsigned short* __restrict__ Wh_bf,
                    const unsigned short* __restrict__ Uh_bf,
                    int son_start, int fa_start, int S, int total, int npt) {
    __shared__ float          s_hh [16 * PF];   // f32 hh rows
    __shared__ unsigned short s_hhb[16 * PB];   // bf16 hh rows
    __shared__ unsigned short s_rhb[16 * PB];   // bf16 (r*hh) rows
    __shared__ int s_node[16];
    __shared__ int s_src0[16];
    __shared__ int s_valid[16];

    const int tid = threadIdx.x;

    // ---- phase 0: node index setup -------------------------------------
    if (tid < 16) {
        const int i   = blockIdx.x * 16 + tid;
        const int val = (i < total) ? 1 : 0;
        const int ii  = val ? i : (total - 1);
        const int tree = ii / S, pos = ii % S;
        int node, src0;
        if (!IS_DOWN) {
            node = tree * npt + fa_start + pos;
            src0 = tree * npt + son_start + 2 * pos;   // child0 (child1 = +1)
        } else {
            node = tree * npt + son_start + pos;
            src0 = tree * npt + fa_start + (pos >> 1); // parent
        }
        s_node[tid] = node; s_src0[tid] = src0; s_valid[tid] = val;
    }
    __syncthreads();

    // ---- phase 1: build hh (f32 + bf16) in LDS --------------------------
    for (int q = tid; q < 16 * 128; q += 256) {
        const int m = q >> 7;
        const int c = (q & 127) << 2;
        const float* p0 = hstate + (long)s_src0[m] * H_DIM + c;
        float4 a = *(const float4*)p0;
        if (!IS_DOWN) {
            float4 b = *(const float4*)(p0 + H_DIM);   // sibling child row
            a.x += b.x; a.y += b.y; a.z += b.z; a.w += b.w;
        }
        float* f = s_hh + m * PF + c;
        f[0] = a.x; f[1] = a.y; f[2] = a.z; f[3] = a.w;
        unsigned short* g = s_hhb + m * PB + c;
        g[0] = f2bf(a.x); g[1] = f2bf(a.y); g[2] = f2bf(a.z); g[3] = f2bf(a.w);
    }
    __syncthreads();

    const int lane = tid & 31, wave = tid >> 5;
    const int lh = lane >> 4, ln = lane & 15;
    const unsigned short* xrow  = xbf + (long)s_node[ln] * H_DIM;
    const unsigned short* hhrow = s_hhb + ln * PB;

    // ---- phase 2: z and r, write r*hh (bf16) to LDS ---------------------
    v8f zkeep[4];
#pragma unroll
    for (int t = 0; t < 4; ++t) {
        const int n0 = wave * 64 + t * 16;
        const unsigned short* wzr = Wz_bf + (long)(n0 + ln) * H_DIM;
        const unsigned short* uzr = Uz_bf + (long)(n0 + ln) * H_DIM;
        const unsigned short* wrr = Wr_bf + (long)(n0 + ln) * H_DIM;
        const unsigned short* urr = Ur_bf + (long)(n0 + ln) * H_DIM;
        v8f za = {}, ra = {};
#pragma unroll
        for (int k0 = 0; k0 < H_DIM; k0 += 32) {
            v16bf ax = load_a(xrow,  k0, lh);
            v16bf ah = load_a(hhrow, k0, lh);
            za = wmma_bf16(ax, load_b(wzr, k0, lh), za);
            za = wmma_bf16(ah, load_b(uzr, k0, lh), za);
            ra = wmma_bf16(ax, load_b(wrr, k0, lh), ra);
            ra = wmma_bf16(ah, load_b(urr, k0, lh), ra);
        }
        const float vbz = bz[n0 + ln];
        const float vbr = br[n0 + ln];
#pragma unroll
        for (int d = 0; d < 8; ++d) {
            const int m = d + 8 * lh;
            const float z  = sigmoidf(za[d] + vbz);
            const float r  = sigmoidf(ra[d] + vbr);
            const float hs = s_hh[m * PF + n0 + ln];
            s_rhb[m * PB + n0 + ln] = f2bf(r * hs);
            za[d] = z;
        }
        zkeep[t] = za;
    }
    __syncthreads();

    // ---- phase 3: cand, blend, store ------------------------------------
    const unsigned short* rhrow = s_rhb + ln * PB;
#pragma unroll
    for (int t = 0; t < 4; ++t) {
        const int n0 = wave * 64 + t * 16;
        const unsigned short* whr = Wh_bf + (long)(n0 + ln) * H_DIM;
        const unsigned short* uhr = Uh_bf + (long)(n0 + ln) * H_DIM;
        v8f ca = {};
#pragma unroll
        for (int k0 = 0; k0 < H_DIM; k0 += 32) {
            ca = wmma_bf16(load_a(xrow,  k0, lh), load_b(whr, k0, lh), ca);
            ca = wmma_bf16(load_a(rhrow, k0, lh), load_b(uhr, k0, lh), ca);
        }
        const v8f za = zkeep[t];
#pragma unroll
        for (int d = 0; d < 8; ++d) {
            const int m = d + 8 * lh;
            const float cand = tanhf(ca[d]);
            const float z    = za[d];
            const float hs   = s_hh[m * PF + n0 + ln];
            const float hv   = z * hs + (1.0f - z) * cand;
            if (s_valid[m])
                hstate[(long)s_node[m] * H_DIM + n0 + ln] = hv;
        }
    }
}

// ---------------------------------------------------------------------------
// hd[roots] = h[roots]
// ---------------------------------------------------------------------------
__global__ void copy_roots(const float* __restrict__ h, float* __restrict__ hd,
                           int root_start, int npt) {
    const long base = (long)(blockIdx.x * npt + root_start) * H_DIM;
    for (int t = threadIdx.x; t < H_DIM; t += blockDim.x)
        hd[base + t] = h[base + t];
}

// ---------------------------------------------------------------------------
// out = h @ W1^T + b1 + hd @ W2^T + b2   (all 32760 nodes)
// ---------------------------------------------------------------------------
__global__ __launch_bounds__(256)
void ffn_out(const float* __restrict__ h, const float* __restrict__ hd,
             const unsigned short* __restrict__ W1bf, const float* __restrict__ b1,
             const unsigned short* __restrict__ W2bf, const float* __restrict__ b2,
             float* __restrict__ out, int ntotal) {
    __shared__ unsigned short s_h[16 * PB];
    __shared__ unsigned short s_d[16 * PB];

    const int tid  = threadIdx.x;
    const int row0 = blockIdx.x * 16;

    for (int q = tid; q < 16 * 128; q += 256) {
        const int m = q >> 7;
        const int c = (q & 127) << 2;
        int node = row0 + m;
        if (node >= ntotal) node = ntotal - 1;
        const float4 a = *(const float4*)(h  + (long)node * H_DIM + c);
        const float4 b = *(const float4*)(hd + (long)node * H_DIM + c);
        unsigned short* p = s_h + m * PB + c;
        p[0] = f2bf(a.x); p[1] = f2bf(a.y); p[2] = f2bf(a.z); p[3] = f2bf(a.w);
        unsigned short* r = s_d + m * PB + c;
        r[0] = f2bf(b.x); r[1] = f2bf(b.y); r[2] = f2bf(b.z); r[3] = f2bf(b.w);
    }
    __syncthreads();

    const int lane = tid & 31, wave = tid >> 5;
    const int lh = lane >> 4, ln = lane & 15;
    const unsigned short* hrow = s_h + ln * PB;
    const unsigned short* drow = s_d + ln * PB;

#pragma unroll
    for (int t = 0; t < 4; ++t) {
        const int n0 = wave * 64 + t * 16;
        const unsigned short* w1 = W1bf + (long)(n0 + ln) * H_DIM;
        const unsigned short* w2 = W2bf + (long)(n0 + ln) * H_DIM;
        v8f acc = {};
#pragma unroll
        for (int k0 = 0; k0 < H_DIM; k0 += 32) {
            acc = wmma_bf16(load_a(hrow, k0, lh), load_b(w1, k0, lh), acc);
            acc = wmma_bf16(load_a(drow, k0, lh), load_b(w2, k0, lh), acc);
        }
        const float bb = b1[n0 + ln] + b2[n0 + ln];
#pragma unroll
        for (int d = 0; d < 8; ++d) {
            const int m    = d + 8 * lh;
            const int node = row0 + m;
            if (node < ntotal)
                out[(long)node * H_DIM + n0 + ln] = acc[d] + bb;
        }
    }
}

// ---------------------------------------------------------------------------
// Host launcher
// ---------------------------------------------------------------------------
extern "C" void kernel_launch(void* const* d_in, const int* in_sizes, int n_in,
                              void* d_out, int out_size, void* d_ws, size_t ws_size,
                              hipStream_t stream) {
    (void)in_sizes; (void)n_in; (void)out_size; (void)ws_size;

    const int NLEAF = 2048, BATCH = 8;
    const int NPT   = 2 * NLEAF - 1;          // 4095
    const int NTOT  = NPT * BATCH;            // 32760

    // input pointers (setup_inputs order)
    const float* feat = (const float*)d_in[0];
    const float* bUz  = (const float*)d_in[6];
    const float* bUr  = (const float*)d_in[8];
    const float* bUzd = (const float*)d_in[14];
    const float* bUrd = (const float*)d_in[16];
    const float* b1   = (const float*)d_in[18];
    const float* b2   = (const float*)d_in[20];

    // workspace carve-out (256B aligned)
    char* w = (char*)d_ws;
    auto carve = [&](size_t bytes) -> void* {
        void* p = (void*)w;
        w += (bytes + 255) & ~(size_t)255;
        return p;
    };
    unsigned short* feat_bf = (unsigned short*)carve((size_t)NTOT * H_DIM * 2);
    float*          hbuf    = (float*)carve((size_t)NTOT * H_DIM * 4);
    float*          hdbuf   = (float*)carve((size_t)NTOT * H_DIM * 4);
    unsigned short* wbf[14];
    for (int i = 0; i < 14; ++i)
        wbf[i] = (unsigned short*)carve((size_t)H_DIM * H_DIM * 2);
    // wbf: 0=Wh 1=Wz 2=Wr 3=Uh 4=Uz 5=Ur 6=Whd 7=Wzd 8=Wrd 9=Uhd 10=Uzd 11=Urd 12=W1 13=W2
    const int widx[14] = {1, 2, 3, 4, 5, 7, 9, 10, 11, 12, 13, 15, 17, 19};

    // converts
    cvt_f32_bf16<<<2048, 256, 0, stream>>>(feat, feat_bf, NTOT * H_DIM);
    for (int i = 0; i < 14; ++i)
        cvt_f32_bf16<<<512, 256, 0, stream>>>((const float*)d_in[widx[i]], wbf[i],
                                              H_DIM * H_DIM);

    // tree level starts (bottom-up node numbering per tree)
    int starts[12];
    {
        int s = 0, sz = NLEAF;
        for (int l = 0; l < 12; ++l) { starts[l] = s; s += sz; sz >>= 1; }
    }

    // leaves
    tree_gru_leaf<<<(NLEAF * BATCH) / 16, 256, 0, stream>>>(
        feat_bf, hbuf, wbf[1], wbf[0], NPT);

    // bottom-up layers
    for (int l = 0; l < 11; ++l) {
        const int S = NLEAF >> (l + 1);
        const int total = S * BATCH;
        const int blocks = (total + 15) / 16;
        tree_gru_layer<false><<<blocks, 256, 0, stream>>>(
            feat_bf, hbuf,
            wbf[1], wbf[4], bUz,
            wbf[2], wbf[5], bUr,
            wbf[0], wbf[3],
            starts[l], starts[l + 1], S, total, NPT);
    }

    // roots: hd = h
    copy_roots<<<BATCH, 256, 0, stream>>>(hbuf, hdbuf, starts[11], NPT);

    // top-down layers
    for (int l = 10; l >= 0; --l) {
        const int S = NLEAF >> l;           // sons at this level
        const int total = S * BATCH;
        const int blocks = (total + 15) / 16;
        tree_gru_layer<true><<<blocks, 256, 0, stream>>>(
            feat_bf, hdbuf,
            wbf[7], wbf[10], bUzd,
            wbf[8], wbf[11], bUrd,
            wbf[6], wbf[9],
            starts[l], starts[l + 1], S, total, NPT);
    }

    // final FFN
    ffn_out<<<(NTOT + 15) / 16, 256, 0, stream>>>(
        hbuf, hdbuf, wbf[12], b1, wbf[13], b2, (float*)d_out, NTOT);
}
